// Net_24172075942219
// MI455X (gfx1250) — matrix operand order repaired and checked by reading, not compiled, not run on hardware
//
#include <hip/hip_runtime.h>
#include <hip/hip_bf16.h>

typedef unsigned short ushort_t;
typedef unsigned int uint_t;

// CDNA5 WMMA vector types (wave32: 16 bf16 = 8 VGPRs, 8 f32 = 8 VGPRs)
typedef __bf16 v16bf __attribute__((ext_vector_type(16)));
typedef float  v8f   __attribute__((ext_vector_type(8)));

// TDM descriptor vector types (6-arg builtin form on this toolchain)
typedef unsigned int u32x4 __attribute__((ext_vector_type(4)));
typedef int          i32x8 __attribute__((ext_vector_type(8)));
typedef int          i32x4 __attribute__((ext_vector_type(4)));

#define N_PAIRS 8192
#define DKEY    1024
#define TWO_D   2048
#define HID     100
#define HPAD    112           // 7 tiles of 16
#define CT_N    7
#define KT_N    64            // 2048 / 32
#define RT_N    512           // 8192 / 16

// bytes of one swizzled B fragment chunk: 32 lanes * 16 bf16 * 2B
#define CHUNK_B 1024
// LDS B stage: 7 ct * 32 kt * 1KB = 224 KB (fits 320 KB WGP LDS)
#define LDSB_BYTES (CT_N * 32 * CHUNK_B)

// workspace layout (bytes)
#define OFF_AS   0ull                                   // 8192*2048 bf16 = 32 MB
#define OFF_W1S  (OFF_AS + (size_t)RT_N*KT_N*32*16*2)   // 7*64*32*16 bf16 = 448 KB
#define OFF_S    (OFF_W1S + (size_t)CT_N*KT_N*32*16*2)  // 8192 f32
#define OFF_ACC  (OFF_S + (size_t)N_PAIRS*4)            // 2 f64

__device__ __forceinline__ ushort_t f32_to_bf16_rne(float f) {
    uint_t u = __float_as_uint(f);
    u += 0x7FFFu + ((u >> 16) & 1u);   // round-to-nearest-even
    return (ushort_t)(u >> 16);
}

__global__ void init_acc_kernel(double* acc) {
    acc[0] = 0.0;
    acc[1] = 0.0;
}

// Convert q||x (f32) into bf16 A fragments, pre-swizzled to the CDNA5
// 16-bit 16x32 A-matrix lane layout: chunk index = ((rt*64+kt)*32+lane),
// lane holds row = rt*16 + (lane&15), k = kt*32 + (lane>>4)*8 + (i&7) + (i>>3)*16.
__global__ void convert_A_kernel(const float* __restrict__ q,
                                 const float* __restrict__ x,
                                 ushort_t* __restrict__ As) {
    uint_t tid  = blockIdx.x * blockDim.x + threadIdx.x;   // 512*64*32 threads
    uint_t lane = tid & 31u;
    uint_t kt   = (tid >> 5) & 63u;
    uint_t rt   = tid >> 11;
    uint_t row  = rt * 16u + (lane & 15u);
    uint_t kb   = kt * 32u + (lane >> 4) * 8u;

    ushort_t frag[16];
#pragma unroll
    for (int i = 0; i < 16; ++i) {
        uint_t k = kb + (uint_t)(i & 7) + (uint_t)((i >> 3) << 4);
        float v = (k < DKEY) ? q[(size_t)row * DKEY + k]
                             : x[(size_t)row * DKEY + (k - DKEY)];
        frag[i] = f32_to_bf16_rne(v);
    }
    uint4* dst = (uint4*)(As + (size_t)tid * 16u);
    dst[0] = ((const uint4*)frag)[0];
    dst[1] = ((const uint4*)frag)[1];
}

// Convert W1 (2048x100 f32, row-major) into bf16 B fragments, padded to 112
// cols, in the CDNA5 16-bit 32x16 B-matrix lane layout:
// chunk = ((ct*64+kt)*32+lane), lane holds col = ct*16+(lane&15),
// k = kt*32 + (lane>>4)*8 + (i&7) + (i>>3)*16.
__global__ void convert_W1_kernel(const float* __restrict__ W1,
                                  ushort_t* __restrict__ W1s) {
    uint_t tid  = blockIdx.x * blockDim.x + threadIdx.x;   // 7*64*32 threads
    if (tid >= (uint_t)(CT_N * KT_N * 32)) return;
    uint_t lane = tid & 31u;
    uint_t kt   = (tid >> 5) & 63u;
    uint_t ct   = tid >> 11;
    uint_t col  = ct * 16u + (lane & 15u);
    uint_t kb   = kt * 32u + (lane >> 4) * 8u;

    ushort_t frag[16];
#pragma unroll
    for (int i = 0; i < 16; ++i) {
        uint_t k = kb + (uint_t)(i & 7) + (uint_t)((i >> 3) << 4);
        float v = (col < HID) ? W1[(size_t)k * HID + col] : 0.0f;
        frag[i] = f32_to_bf16_rne(v);
    }
    uint4* dst = (uint4*)(W1s + (size_t)tid * 16u);
    dst[0] = ((const uint4*)frag)[0];
    dst[1] = ((const uint4*)frag)[1];
}

// Issue one 2D TDM copy: 7 rows (ct tiles) x 32KB of swizzled W1 fragments,
// row stride 64KB, into LDS at lds_off. data_size = 8B.
__device__ __forceinline__ void tdm_load_w1_half(const ushort_t* W1s, int half,
                                                 uint_t lds_off) {
    unsigned long long ga = (unsigned long long)(const void*)W1s
                          + (unsigned long long)half * 32u * CHUNK_B;
    u32x4 g0;
    g0[0] = 1u;                                  // count=1, is_restore=0
    g0[1] = lds_off;                             // lds_addr (bytes)
    g0[2] = (uint_t)(ga & 0xFFFFFFFFull);        // global_addr[31:0]
    g0[3] = (uint_t)((ga >> 32) & 0x01FFFFFFull) // global_addr[56:32]
          | (2u << 30);                          // type=2 (image/tensor)

    i32x8 g1;
    g1[0] = (int)(3u << 16);        // wg_mask=0, data_size=3 (8B), no flags
    g1[1] = (int)(4096u << 16);     // tensor_dim0[15:0]=4096 (in 8B units)
    g1[2] = (int)(7u << 16);        // tensor_dim0[31:16]=0 | tensor_dim1=7
    g1[3] = (int)(4096u << 16);     // tensor_dim1[31:16]=0 | tile_dim0=4096
    g1[4] = 7;                      // tile_dim1=7, tile_dim2=0 (2D tile)
    g1[5] = 8192;                   // tensor_dim0_stride = 8192 (64KB/8B)
    g1[6] = 0;                      // stride hi / dim1_stride lo
    g1[7] = 0;

    i32x4 gz4 = (i32x4){0, 0, 0, 0};               // groups 2/3 unused (2D)
    i32x8 gz8 = (i32x8){0, 0, 0, 0, 0, 0, 0, 0};   // extra group (unused)

    __builtin_amdgcn_tensor_load_to_lds(g0, g1, gz4, gz4, gz8, 0);
    __builtin_amdgcn_s_wait_tensorcnt(0);
}

// Fused encoder: per wave, one 16-row tile. B operand staged in LDS by the
// Tensor Data Mover (one 224KB half-K tile at a time), A streamed from global.
// Epilogue applies bias+ReLU, dots with W2, reduces across 16-lane halves and
// writes s directly (h never hits memory).
__global__ void __launch_bounds__(256)
encoder_gemm_kernel(const ushort_t* __restrict__ As,
                    const ushort_t* __restrict__ W1s,
                    const float* __restrict__ b1,
                    const float* __restrict__ W2,
                    const float* __restrict__ b2,
                    float* __restrict__ sbuf) {
    extern __shared__ ushort_t ldsb[];   // CT_N*32 chunks * 32 lanes * 16 bf16

    const uint_t lane = threadIdx.x & 31u;
    const uint_t wave = threadIdx.x >> 5;
    const uint_t rt   = blockIdx.x * 8u + wave;           // 64 blocks * 8 waves

    const v16bf* Av = (const v16bf*)As + (size_t)rt * (KT_N * 32) + lane;
    const v16bf* Bl = (const v16bf*)ldsb + lane;

    v8f acc[CT_N];
#pragma unroll
    for (int ct = 0; ct < CT_N; ++ct) acc[ct] = (v8f){0,0,0,0,0,0,0,0};

    for (int half = 0; half < 2; ++half) {
        if (half) __syncthreads();       // protect LDS reuse across halves
        if (wave == 0) {
            tdm_load_w1_half(W1s, half,
                             (uint_t)(unsigned long long)(const void*)ldsb);
        }
        __syncthreads();                 // TDM data visible to all waves

        for (int ktl = 0; ktl < 32; ++ktl) {
            int kt = half * 32 + ktl;
            v16bf a = Av[(size_t)kt * 32];
            if (kt + 1 < KT_N)
                __builtin_prefetch((const void*)&Av[(size_t)(kt + 1) * 32], 0, 0);
#pragma unroll
            for (int ct = 0; ct < CT_N; ++ct) {
                v16bf bm = Bl[(size_t)(ct * 32 + ktl) * 32];
                acc[ct] = __builtin_amdgcn_wmma_f32_16x16x32_bf16(
                    false, a, false, bm, (short)0, acc[ct], false, false);
            }
        }
    }

    // Epilogue: C layout — lane holds col = ct*16+(lane&15); acc element r is
    // row rt*16 + r (lanes 0-15) or rt*16 + 8 + r (lanes 16-31).
    const uint_t nlo = lane & 15u;
    float p[8];
#pragma unroll
    for (int r = 0; r < 8; ++r) p[r] = 0.0f;

#pragma unroll
    for (int ct = 0; ct < CT_N; ++ct) {
        uint_t col = (uint_t)ct * 16u + nlo;
        float w2 = (col < HID) ? W2[col] : 0.0f;
        float bb = (col < HID) ? b1[col] : 0.0f;
#pragma unroll
        for (int r = 0; r < 8; ++r) {
            float h = acc[ct][r] + bb;
            h = fmaxf(h, 0.0f);
            p[r] = fmaf(h, w2, p[r]);
        }
    }

#pragma unroll
    for (int m = 1; m < 16; m <<= 1) {
#pragma unroll
        for (int r = 0; r < 8; ++r) p[r] += __shfl_xor(p[r], m, 32);
    }

    if ((lane & 15u) == 0u) {
        float b2v = b2[0];
        uint_t rbase = rt * 16u + ((lane >> 4) ? 8u : 0u);
#pragma unroll
        for (int r = 0; r < 8; ++r) sbuf[rbase + r] = p[r] + b2v;
    }
}

// Tiled O(N^2) masked-pair BCE. 256x256 tiles; blocks strictly below the
// diagonal are pruned. j-tile staged in LDS; f64 atomics for the masked mean.
__global__ void __launch_bounds__(256)
pair_loss_kernel(const int* __restrict__ b, const int* __restrict__ y,
                 const float* __restrict__ s, double* __restrict__ acc) {
    if (blockIdx.x > blockIdx.y) return;   // need i < j somewhere in tile

    __shared__ float sj[256];
    __shared__ int   bj[256];
    __shared__ int   yj[256];
    __shared__ float rsum[256];
    __shared__ int   rcnt[256];

    const int t  = threadIdx.x;
    const int j0 = blockIdx.y * 256;
    sj[t] = s[j0 + t];
    bj[t] = b[j0 + t];
    yj[t] = y[j0 + t];
    __syncthreads();

    const int i  = blockIdx.x * 256 + t;
    const float si = s[i];
    const int   bi = b[i];
    const int   yi = y[i];
    const float z  = (float)yi;

    float lsum = 0.0f;
    int   lcnt = 0;
    for (int jj = 0; jj < 256; ++jj) {
        int j = j0 + jj;
        if ((bi == bj[jj]) & (yi != yj[jj]) & (i < j)) {
            float d = si - sj[jj];
            float bce = fmaxf(d, 0.0f) - d * z + log1pf(expf(-fabsf(d)));
            lsum += bce;
            lcnt += 1;
        }
    }

    rsum[t] = lsum;
    rcnt[t] = lcnt;
    __syncthreads();
    for (int off = 128; off > 0; off >>= 1) {
        if (t < off) { rsum[t] += rsum[t + off]; rcnt[t] += rcnt[t + off]; }
        __syncthreads();
    }
    if (t == 0) {
        atomicAdd(&acc[0], (double)rsum[0]);
        atomicAdd(&acc[1], (double)rcnt[0]);
    }
}

__global__ void finalize_kernel(const double* acc, float* out) {
    double c = acc[1];
    out[0] = (c > 0.0) ? (float)(acc[0] / c) : 0.0f;
}

extern "C" void kernel_launch(void* const* d_in, const int* in_sizes, int n_in,
                              void* d_out, int out_size, void* d_ws, size_t ws_size,
                              hipStream_t stream) {
    const int*   b  = (const int*)d_in[0];
    const float* q  = (const float*)d_in[1];
    const float* x  = (const float*)d_in[2];
    const int*   y  = (const int*)d_in[3];
    const float* W1 = (const float*)d_in[4];
    const float* b1 = (const float*)d_in[5];
    const float* W2 = (const float*)d_in[6];
    const float* b2 = (const float*)d_in[7];
    float* out = (float*)d_out;

    char* w = (char*)d_ws;
    ushort_t* As   = (ushort_t*)(w + OFF_AS);
    ushort_t* W1s  = (ushort_t*)(w + OFF_W1S);
    float*    sbuf = (float*)(w + OFF_S);
    double*   acc  = (double*)(w + OFF_ACC);

    init_acc_kernel<<<1, 1, 0, stream>>>(acc);

    // 512 rt * 64 kt * 32 lanes = 1,048,576 threads
    convert_A_kernel<<<4096, 256, 0, stream>>>(q, x, As);

    // 7 ct * 64 kt * 32 lanes = 14,336 threads
    convert_W1_kernel<<<56, 256, 0, stream>>>(W1, W1s);

    // 64 blocks * 8 waves = 512 row tiles; 224KB dynamic LDS for the B stage
    encoder_gemm_kernel<<<64, 256, LDSB_BYTES, stream>>>(As, W1s, b1, W2, b2, sbuf);

    pair_loss_kernel<<<dim3(32, 32), 256, 0, stream>>>(b, y, sbuf, acc);

    finalize_kernel<<<1, 1, 0, stream>>>(acc, out);
}
